// GraphGPSBlock_90151363543201
// MI455X (gfx1250) — matrix operand order repaired and checked by reading, not compiled, not run on hardware
//
#include <hip/hip_runtime.h>

typedef __attribute__((ext_vector_type(16))) _Float16 v16h;
typedef __attribute__((ext_vector_type(8)))  float    v8f;

#define N_NODES 4096
#define N_EDGES 16384

// ---------------- workspace layout (bytes) ----------------
#define OFF_BUFA 0ULL                  // 67 MB : pooled_pos, later x1 (16ch)
#define OFF_BUFB 67108864ULL           // 67 MB : pooled_neg, later refine r1
#define OFF_BUFC 134217728ULL          // 134 MB: 32ch conv intermediates
#define OFF_TAIL 268435456ULL
#define OFF_GID     (OFF_TAIL + 0ULL)          // 4096 int
#define OFF_POOLED  (OFF_TAIL + 16384ULL)      // 4096x16 f32
#define OFF_Q       (OFF_TAIL + 278528ULL)
#define OFF_K       (OFF_TAIL + 540672ULL)
#define OFF_V       (OFF_TAIL + 802816ULL)
#define OFF_O       (OFF_TAIL + 1064960ULL)
#define OFF_TOK     (OFF_TAIL + 1327104ULL)
#define OFF_DELTA   (OFF_TAIL + 1589248ULL)
#define OFF_WT      (OFF_TAIL + 1851392ULL)    // f16 pre-swizzled weight fragments

// halfword offsets inside WT
#define W_WB1 0        // conv1: 14 chunks * NT2 * 512
#define W_WB2 14336    // conv2: 9 chunks * NT2
#define W_WB3 23552    // conv3: 9 chunks * NT1
#define W_RB1 28160    // ref1 : 5 chunks * NT1
#define W_RB2 30720
#define W_WQ  33280    // 1 tile (512)
#define W_WK  33792
#define W_WV  34304
#define W_WO  34816
#define W_F1  35328    // 4 tiles
#define W_F2  37376    // 2 k-chunks
#define W_TOTAL 38400

// ---------------- misc kernels ----------------
__global__ __launch_bounds__(256) void zero2_kernel(float* a, float* b, int n) {
  int i = blockIdx.x * 256 + threadIdx.x;
  if (i < n) { a[i] = 0.f; b[i] = 0.f; }
}

__global__ __launch_bounds__(256) void comp_init_kernel(int* lbl) {
  int i = blockIdx.x * 256 + threadIdx.x;
  if (i < N_NODES) lbl[i] = i;
}

__global__ __launch_bounds__(256) void comp_edge_kernel(const int* __restrict__ edges, int* lbl) {
  int e = blockIdx.x * 256 + threadIdx.x;
  if (e >= N_EDGES) return;
  int s = edges[3 * e + 0];
  int d = edges[3 * e + 2];
  int ls = lbl[s], ld = lbl[d];
  int m = ls < ld ? ls : ld;
  atomicMin(&lbl[s], m);
  atomicMin(&lbl[d], m);
}

__global__ __launch_bounds__(256) void comp_jump_kernel(int* lbl) {
  int i = blockIdx.x * 256 + threadIdx.x;
  if (i < N_NODES) lbl[i] = lbl[lbl[i]];
}

__global__ __launch_bounds__(256) void edge_pool_kernel(const int* __restrict__ edges,
                                                        const float* __restrict__ x,
                                                        float* bufPos, float* bufNeg) {
  int e = blockIdx.x;
  int rel = edges[3 * e + 1];
  if (rel == 0) return;
  float* dstbuf = (rel > 0) ? bufPos : bufNeg;
  int s = edges[3 * e + 0];
  int d = edges[3 * e + 2];
  const float* xs = x + (size_t)s * 4096;
  const float* xd = x + (size_t)d * 4096;
  float* ps = dstbuf + (size_t)s * 4096;
  float* pd = dstbuf + (size_t)d * 4096;
  for (int i = threadIdx.x; i < 4096; i += 256) {
    unsafeAtomicAdd(&pd[i], xs[i]);
    unsafeAtomicAdd(&ps[i], xd[i]);
  }
}

// ---------------- weight pre-swizzle ----------------
// Conv weights -> exact per-lane A-fragment order (16x32 f16 A layout):
// index = ((kc*NT + nt)*32 + lane)*16 + ii ; lane: M=lane%16 (=oc in tile), half=lane/16
// a[ii] <-> K = kc*32 + (ii<8 ? half*8+ii : 16+half*8+(ii-8))
__device__ inline _Float16 swzW(const float* __restrict__ w, int i, int NT, int KREAL) {
  int ii = i & 15;
  int lane = (i >> 4) & 31;
  int t = i >> 9;            // kc*NT + nt
  int nt = t % NT;
  int kc = t / NT;
  int half = lane >> 4;
  int kk = kc * 32 + ((ii < 8) ? (half * 8 + ii) : (16 + half * 8 + (ii - 8)));
  int oc = nt * 16 + (lane & 15);
  return (_Float16)((kk < KREAL) ? w[oc * KREAL + kk] : 0.f);
}

// Small GEMM weights -> per-lane B-fragment order (32x16 f16 B layout):
// index = lane*16 + ii ; lane: N=lane%16, half=lane/16 ; b[ii] <-> K = (ii<8 ? half*8+ii : 16+half*8+(ii-8))
// B[k][o] = W[o][k] (W row-major [O][KIN]), K beyond KIN zero-padded.
__device__ inline _Float16 swzB(const float* __restrict__ w, int lane, int ii, int KIN, int obase) {
  int half = lane >> 4, l15 = lane & 15;
  int kk = (ii < 8) ? (half * 8 + ii) : (16 + half * 8 + (ii - 8));
  return (_Float16)((kk < KIN) ? w[(obase + l15) * KIN + kk] : 0.f);
}

__global__ __launch_bounds__(256) void prep_kernel(
    const float* w1, const float* w2, const float* w3,
    const float* rw1, const float* rw2,
    const float* wq, const float* wk, const float* wv, const float* wo,
    const float* f1, const float* f2, _Float16* wt) {
  int i = blockIdx.x * 256 + threadIdx.x;
  if (i < 14336) { wt[W_WB1 + i] = swzW(w1, i, 2, 432); return; }
  i -= 14336;
  if (i < 9216)  { wt[W_WB2 + i] = swzW(w2, i, 2, 288); return; }
  i -= 9216;
  if (i < 4608)  { wt[W_WB3 + i] = swzW(w3, i, 1, 288); return; }
  i -= 4608;
  if (i < 2560)  { wt[W_RB1 + i] = swzW(rw1, i, 1, 144); return; }
  i -= 2560;
  if (i < 2560)  { wt[W_RB2 + i] = swzW(rw2, i, 1, 144); return; }
  i -= 2560;
  if (i < 512)   { wt[W_WQ + i] = swzB(wq, i >> 4, i & 15, 16, 0); return; }
  i -= 512;
  if (i < 512)   { wt[W_WK + i] = swzB(wk, i >> 4, i & 15, 16, 0); return; }
  i -= 512;
  if (i < 512)   { wt[W_WV + i] = swzB(wv, i >> 4, i & 15, 16, 0); return; }
  i -= 512;
  if (i < 512)   { wt[W_WO + i] = swzB(wo, i >> 4, i & 15, 16, 0); return; }
  i -= 512;
  if (i < 2048)  { int nt = i >> 9; int r = i & 511;              // ffn1: K=16 pad 32, O=64 in 4 tiles
                   wt[W_F1 + i] = swzB(f1, r >> 4, r & 15, 16, nt * 16); return; }
  i -= 2048;
  if (i < 1024)  { int kc = i >> 9; int lane = (i >> 4) & 31; int ii = i & 15;  // ffn2: K=64, O=16
                   int k = kc * 32 + ((ii < 8) ? ((lane >> 4) * 8 + ii) : (16 + (lane >> 4) * 8 + (ii - 8)));
                   int o = lane & 15;
                   wt[W_F2 + i] = (_Float16)(f2[o * 64 + k]); return; }
}

// ---------------- WMMA implicit-GEMM conv ----------------
// A = weights (pre-swizzled, loaded b128 from global), B = im2col patch gathered from
// zero-padded 18x18 f16 LDS image via precomputed offset table. One node per workgroup,
// 8 waves; wave w owns output rows {w, w+8} x all N-tiles (acc[2][NT]).
template <int CIN, int COUT, int KPAD, bool SPLIT3, bool RESID>
__global__ __launch_bounds__(256) void conv_kernel(const float* __restrict__ in0,
                                                   const float* __restrict__ in1,
                                                   const float* __restrict__ in2,
                                                   const _Float16* __restrict__ wA,
                                                   const float* __restrict__ bias,
                                                   const float* __restrict__ resid,
                                                   float* __restrict__ out) {
  constexpr int NT = COUT / 16;
  constexpr int KREAL = CIN * 9;
  constexpr int NCHUNK = KPAD / 32;
  __shared__ _Float16 s_in[CIN * 324];
  __shared__ int s_off[KPAD];
  const int node = blockIdx.x;
  const int tid = threadIdx.x;

  for (int i = tid; i < CIN * 324; i += 256) s_in[i] = (_Float16)0.f;
  for (int i = tid; i < KPAD; i += 256) {
    int kk = (i < KREAL) ? i : (KREAL - 1);  // clamped; padded K has zero A rows
    int c = kk / 9, r = kk - c * 9;
    int dy = r / 3, dx = r - dy * 3;
    s_off[i] = c * 324 + dy * 18 + dx;
  }
  __syncthreads();
  for (int i = tid; i < CIN * 256; i += 256) {
    int c = i >> 8, p = i & 255, y = p >> 4, x = p & 15;
    float v;
    if constexpr (SPLIT3) {
      const float* src = (c < 16) ? in0 : ((c < 32) ? in1 : in2);
      v = src[((size_t)node * 16 + (c & 15)) * 256 + p];
    } else {
      v = in0[((size_t)node * CIN + c) * 256 + p];
    }
    s_in[c * 324 + (y + 1) * 18 + (x + 1)] = (_Float16)v;
  }
  __syncthreads();

  const int wave = tid >> 5, lane = tid & 31;
  const int l15 = lane & 15, half = lane >> 4;

  v8f acc[2][NT] = {};

  for (int kc = 0; kc < NCHUNK; kc++) {
    int k0 = kc * 32;
    int offr[16];
#pragma unroll
    for (int i = 0; i < 16; i++) offr[i] = s_off[k0 + ((i < 8) ? (half * 8 + i) : (16 + half * 8 + (i - 8)))];
    v16h afr[NT];
#pragma unroll
    for (int nt = 0; nt < NT; nt++)
      afr[nt] = *(const v16h*)(wA + ((size_t)(kc * NT + nt) * 32 + lane) * 16);
#pragma unroll
    for (int yy = 0; yy < 2; yy++) {
      int y = wave + yy * 8;
      v16h b;
#pragma unroll
      for (int i = 0; i < 16; i++) b[i] = s_in[offr[i] + y * 18 + l15];
#pragma unroll
      for (int nt = 0; nt < NT; nt++)
        acc[yy][nt] = __builtin_amdgcn_wmma_f32_16x16x32_f16(false, afr[nt], false, b,
                                                             (short)0, acc[yy][nt], false, false);
    }
  }

#pragma unroll
  for (int yy = 0; yy < 2; yy++) {
    int y = wave + yy * 8;
#pragma unroll
    for (int nt = 0; nt < NT; nt++) {
#pragma unroll
      for (int j = 0; j < 8; j++) {
        int oc = nt * 16 + j + 8 * half;
        float v = acc[yy][nt][j] + bias[oc];
        v = (v >= 0.f) ? v : 0.1f * v;  // leaky ReLU
        size_t idx = ((size_t)node * COUT + oc) * 256 + (size_t)y * 16 + l15;
        if constexpr (RESID) v += resid[idx];
        out[idx] = v;
      }
    }
  }
}

// ---------------- spatial mean pool ----------------
__global__ __launch_bounds__(256) void pooled_mean_kernel(const float* __restrict__ x1,
                                                          float* __restrict__ pooled) {
  __shared__ float part[256];
  int n = blockIdx.x, t = threadIdx.x;
  int c = t >> 4;
  float s = 0.f;
  for (int p = (t & 15); p < 256; p += 16) s += x1[((size_t)n * 16 + c) * 256 + p];
  part[t] = s;
  __syncthreads();
  if ((t & 15) == 0) {
    float tot = 0.f;
#pragma unroll
    for (int i = 0; i < 16; i++) tot += part[t + i];
    pooled[n * 16 + c] = tot * (1.f / 256.f);
  }
}

// ---------------- small WMMA GEMM (pre-swizzled B: one contiguous v16h per lane) ----
__device__ inline v8f gemm16_swz(const v16h& a, const _Float16* __restrict__ wBswz, int lane) {
  v16h b = *(const v16h*)(wBswz + lane * 16);
  v8f acc = {};
  return __builtin_amdgcn_wmma_f32_16x16x32_f16(false, a, false, b, (short)0, acc, false, false);
}

// LN1 + Q/K/V projections, one wave per 16-node tile
__global__ __launch_bounds__(32) void qkv_kernel(const float* __restrict__ pooled,
                                                 const float* __restrict__ g1, const float* __restrict__ b1,
                                                 const _Float16* wqB, const _Float16* wkB, const _Float16* wvB,
                                                 const float* bq, const float* bk, const float* bv,
                                                 float* q, float* k, float* v) {
  __shared__ float pt[256];
  __shared__ _Float16 a_lds[16 * 32];
  int n0 = blockIdx.x * 16, tid = threadIdx.x;
  for (int i = tid; i < 256; i += 32) pt[i] = pooled[n0 * 16 + i];
  __syncthreads();
  if (tid < 16) {
    float mu = 0.f;
    for (int c = 0; c < 16; c++) mu += pt[tid * 16 + c];
    mu *= (1.f / 16.f);
    float var = 0.f;
    for (int c = 0; c < 16; c++) { float d = pt[tid * 16 + c] - mu; var += d * d; }
    var *= (1.f / 16.f);
    float rs = rsqrtf(var + 1e-5f);
    for (int c = 0; c < 16; c++) {
      a_lds[tid * 32 + c] = (_Float16)((pt[tid * 16 + c] - mu) * rs * g1[c] + b1[c]);
      a_lds[tid * 32 + 16 + c] = (_Float16)0.f;
    }
  }
  __syncthreads();
  int l15 = tid & 15, half = tid >> 4;
  v16h a;
#pragma unroll
  for (int i = 0; i < 16; i++) {
    int kk = (i < 8) ? (half * 8 + i) : (16 + half * 8 + (i - 8));
    a[i] = a_lds[l15 * 32 + kk];
  }
  v8f cq = gemm16_swz(a, wqB, tid);
  v8f ck = gemm16_swz(a, wkB, tid);
  v8f cv = gemm16_swz(a, wvB, tid);
  float vbq = bq[l15], vbk = bk[l15], vbv = bv[l15];
#pragma unroll
  for (int j = 0; j < 8; j++) {
    int node = n0 + j + 8 * half;
    q[node * 16 + l15] = cq[j] + vbq;
    k[node * 16 + l15] = ck[j] + vbk;
    v[node * 16 + l15] = cv[j] + vbv;
  }
}

// fused masked attention with online softmax (head_dim=4)
__global__ __launch_bounds__(256) void attn_kernel(const float* __restrict__ q,
                                                   const float* __restrict__ k,
                                                   const float* __restrict__ v,
                                                   const int* __restrict__ gid,
                                                   float* __restrict__ o) {
  int h = blockIdx.y;
  int n = blockIdx.x * 256 + threadIdx.x;
  int gn = gid[n];
  const float* qp = q + (size_t)n * 16 + h * 4;
  float q0 = qp[0], q1 = qp[1], q2 = qp[2], q3 = qp[3];
  float mmax = -1e30f, l = 0.f;
  float a0 = 0.f, a1 = 0.f, a2 = 0.f, a3 = 0.f;
  for (int m = 0; m < N_NODES; m++) {
    if (gid[m] != gn) continue;
    const float* kp = k + (size_t)m * 16 + h * 4;
    float s = 0.5f * (q0 * kp[0] + q1 * kp[1] + q2 * kp[2] + q3 * kp[3]);
    if (s > mmax) {
      float sc = __expf(mmax - s);
      l *= sc; a0 *= sc; a1 *= sc; a2 *= sc; a3 *= sc;
      mmax = s;
    }
    float w = __expf(s - mmax);
    l += w;
    const float* vp = v + (size_t)m * 16 + h * 4;
    a0 += w * vp[0]; a1 += w * vp[1]; a2 += w * vp[2]; a3 += w * vp[3];
  }
  float inv = 1.f / l;
  float* op = o + (size_t)n * 16 + h * 4;
  op[0] = a0 * inv; op[1] = a1 * inv; op[2] = a2 * inv; op[3] = a3 * inv;
}

// tokens = pooled + o @ wo^T + bo
__global__ __launch_bounds__(32) void proj_kernel(const float* __restrict__ o,
                                                  const float* __restrict__ pooled,
                                                  const _Float16* woB, const float* bo,
                                                  float* __restrict__ tokens) {
  __shared__ _Float16 a_lds[16 * 32];
  int n0 = blockIdx.x * 16, tid = threadIdx.x;
  for (int i = tid; i < 256; i += 32) {
    a_lds[(i >> 4) * 32 + (i & 15)] = (_Float16)o[n0 * 16 + i];
    a_lds[(i >> 4) * 32 + 16 + (i & 15)] = (_Float16)0.f;
  }
  __syncthreads();
  int l15 = tid & 15, half = tid >> 4;
  v16h a;
#pragma unroll
  for (int i = 0; i < 16; i++) {
    int kk = (i < 8) ? (half * 8 + i) : (16 + half * 8 + (i - 8));
    a[i] = a_lds[l15 * 32 + kk];
  }
  v8f acc = gemm16_swz(a, woB, tid);
  float vb = bo[l15];
#pragma unroll
  for (int j = 0; j < 8; j++) {
    int node = n0 + j + 8 * half;
    tokens[node * 16 + l15] = pooled[node * 16 + l15] + acc[j] + vb;
  }
}

// LN2 + FFN (16->64 GELU 64->16) + residual; writes delta = tokens_new - pooled
__global__ __launch_bounds__(32) void ffn_kernel(const float* __restrict__ tokens,
                                                 const float* __restrict__ pooled,
                                                 const float* __restrict__ g2, const float* __restrict__ b2,
                                                 const _Float16* f1B, const float* fb1,
                                                 const _Float16* f2B, const float* fb2,
                                                 float* __restrict__ delta) {
  __shared__ float tok[256];
  __shared__ _Float16 a_lds[16 * 32];
  __shared__ _Float16 h_lds[16 * 64];
  int n0 = blockIdx.x * 16, tid = threadIdx.x;
  for (int i = tid; i < 256; i += 32) tok[i] = tokens[n0 * 16 + i];
  __syncthreads();
  if (tid < 16) {
    float mu = 0.f;
    for (int c = 0; c < 16; c++) mu += tok[tid * 16 + c];
    mu *= (1.f / 16.f);
    float var = 0.f;
    for (int c = 0; c < 16; c++) { float d = tok[tid * 16 + c] - mu; var += d * d; }
    var *= (1.f / 16.f);
    float rs = rsqrtf(var + 1e-5f);
    for (int c = 0; c < 16; c++) {
      a_lds[tid * 32 + c] = (_Float16)((tok[tid * 16 + c] - mu) * rs * g2[c] + b2[c]);
      a_lds[tid * 32 + 16 + c] = (_Float16)0.f;
    }
  }
  __syncthreads();
  int l15 = tid & 15, half = tid >> 4;
  v16h a;
#pragma unroll
  for (int i = 0; i < 16; i++) {
    int kk = (i < 8) ? (half * 8 + i) : (16 + half * 8 + (i - 8));
    a[i] = a_lds[l15 * 32 + kk];
  }
  // h = gelu(a @ W1^T + b1), 4 N-tiles
#pragma unroll
  for (int nt = 0; nt < 4; nt++) {
    v16h b = *(const v16h*)(f1B + ((size_t)nt * 32 + tid) * 16);
    v8f acc = {};
    acc = __builtin_amdgcn_wmma_f32_16x16x32_f16(false, a, false, b, (short)0, acc, false, false);
    int u = nt * 16 + l15;
    float bvh = fb1[u];
#pragma unroll
    for (int j = 0; j < 8; j++) {
      int node = j + 8 * half;
      float x = acc[j] + bvh;
      float gelu = 0.5f * x * (1.f + erff(x * 0.70710678f));
      h_lds[node * 64 + u] = (_Float16)gelu;
    }
  }
  __syncthreads();
  // f = h @ W2^T + b2 (K=64 in two chunks)
  v8f acc2 = {};
#pragma unroll
  for (int kc = 0; kc < 2; kc++) {
    int k0 = kc * 32;
    v16h a2;
#pragma unroll
    for (int i = 0; i < 16; i++) {
      int kk = k0 + ((i < 8) ? (half * 8 + i) : (16 + half * 8 + (i - 8)));
      a2[i] = h_lds[l15 * 64 + kk];
    }
    v16h b = *(const v16h*)(f2B + ((size_t)kc * 32 + tid) * 16);
    acc2 = __builtin_amdgcn_wmma_f32_16x16x32_f16(false, a2, false, b, (short)0, acc2, false, false);
  }
  float vb = fb2[l15];
#pragma unroll
  for (int j = 0; j < 8; j++) {
    int node = j + 8 * half;
    float tnew = tok[node * 16 + l15] + acc2[j] + vb;
    delta[(size_t)(n0 + node) * 16 + l15] = tnew - pooled[(size_t)(n0 + node) * 16 + l15];
  }
}

// x2 = x1 + delta broadcast over spatial
__global__ __launch_bounds__(256) void broadcast_kernel(const float* __restrict__ x1,
                                                        const float* __restrict__ delta,
                                                        float* __restrict__ out) {
  size_t idx = (size_t)blockIdx.x * 256 + threadIdx.x;
  int n = (int)(idx >> 12);
  int c = (int)((idx >> 8) & 15);
  out[idx] = x1[idx] + delta[n * 16 + c];
}

// ---------------- launcher ----------------
extern "C" void kernel_launch(void* const* d_in, const int* in_sizes, int n_in,
                              void* d_out, int out_size, void* d_ws, size_t ws_size,
                              hipStream_t stream) {
  const float* x      = (const float*)d_in[0];
  const int*   edges  = (const int*)d_in[1];
  const float* cmp_w1 = (const float*)d_in[2];
  const float* cmp_b1 = (const float*)d_in[3];
  const float* cmp_w2 = (const float*)d_in[4];
  const float* cmp_b2 = (const float*)d_in[5];
  const float* cmp_w3 = (const float*)d_in[6];
  const float* cmp_b3 = (const float*)d_in[7];
  const float* ln1_g  = (const float*)d_in[8];
  const float* ln1_b  = (const float*)d_in[9];
  const float* ln2_g  = (const float*)d_in[10];
  const float* ln2_b  = (const float*)d_in[11];
  const float* wq     = (const float*)d_in[12];
  const float* bq     = (const float*)d_in[13];
  const float* wk     = (const float*)d_in[14];
  const float* bk     = (const float*)d_in[15];
  const float* wv     = (const float*)d_in[16];
  const float* bv     = (const float*)d_in[17];
  const float* wo     = (const float*)d_in[18];
  const float* bo     = (const float*)d_in[19];
  const float* ffn_w1 = (const float*)d_in[20];
  const float* ffn_b1 = (const float*)d_in[21];
  const float* ffn_w2 = (const float*)d_in[22];
  const float* ffn_b2 = (const float*)d_in[23];
  const float* ref_w1 = (const float*)d_in[24];
  const float* ref_b1 = (const float*)d_in[25];
  const float* ref_w2 = (const float*)d_in[26];
  const float* ref_b2 = (const float*)d_in[27];

  char* ws = (char*)d_ws;
  float* bufA   = (float*)(ws + OFF_BUFA);
  float* bufB   = (float*)(ws + OFF_BUFB);
  float* bufC   = (float*)(ws + OFF_BUFC);
  int*   gid    = (int*)(ws + OFF_GID);
  float* pooled = (float*)(ws + OFF_POOLED);
  float* qb     = (float*)(ws + OFF_Q);
  float* kb     = (float*)(ws + OFF_K);
  float* vb     = (float*)(ws + OFF_V);
  float* ob     = (float*)(ws + OFF_O);
  float* tok    = (float*)(ws + OFF_TOK);
  float* delta  = (float*)(ws + OFF_DELTA);
  _Float16* wt  = (_Float16*)(ws + OFF_WT);
  float* outp   = (float*)d_out;

  // 1) zero pooled_pos/neg; 2) pre-swizzle weights to f16 fragments
  zero2_kernel<<<65536, 256, 0, stream>>>(bufA, bufB, 16777216);
  prep_kernel<<<(W_TOTAL + 255) / 256, 256, 0, stream>>>(cmp_w1, cmp_w2, cmp_w3, ref_w1, ref_w2,
                                                         wq, wk, wv, wo, ffn_w1, ffn_w2, wt);

  // 3) connected components (min-label propagation + pointer jumping)
  comp_init_kernel<<<16, 256, 0, stream>>>(gid);
  for (int it = 0; it < 24; ++it) {
    comp_edge_kernel<<<64, 256, 0, stream>>>(edges, gid);
    comp_jump_kernel<<<16, 256, 0, stream>>>(gid);
  }

  // 4) signed-edge pooling
  edge_pool_kernel<<<N_EDGES, 256, 0, stream>>>(edges, x, bufA, bufB);

  // 5) compression convs (WMMA implicit GEMM)
  conv_kernel<48, 32, 448, true, false><<<N_NODES, 256, 0, stream>>>(x, bufA, bufB, wt + W_WB1, cmp_b1, nullptr, bufC);
  conv_kernel<32, 32, 288, false, false><<<N_NODES, 256, 0, stream>>>(bufC, nullptr, nullptr, wt + W_WB2, cmp_b2, nullptr, bufC);
  conv_kernel<32, 16, 288, false, true><<<N_NODES, 256, 0, stream>>>(bufC, nullptr, nullptr, wt + W_WB3, cmp_b3, x, bufA);

  // 6) pool + transformer
  pooled_mean_kernel<<<N_NODES, 256, 0, stream>>>(bufA, pooled);
  qkv_kernel<<<N_NODES / 16, 32, 0, stream>>>(pooled, ln1_g, ln1_b,
                                              wt + W_WQ, wt + W_WK, wt + W_WV,
                                              bq, bk, bv, qb, kb, vb);
  attn_kernel<<<dim3(16, 4), 256, 0, stream>>>(qb, kb, vb, gid, ob);
  proj_kernel<<<N_NODES / 16, 32, 0, stream>>>(ob, pooled, wt + W_WO, bo, tok);
  ffn_kernel<<<N_NODES / 16, 32, 0, stream>>>(tok, pooled, ln2_g, ln2_b,
                                              wt + W_F1, ffn_b1, wt + W_F2, ffn_b2, delta);

  // 7) broadcast delta, refinement convs
  broadcast_kernel<<<65536, 256, 0, stream>>>(bufA, delta, outp);
  conv_kernel<16, 16, 160, false, false><<<N_NODES, 256, 0, stream>>>(outp, nullptr, nullptr, wt + W_RB1, ref_b1, nullptr, bufB);
  conv_kernel<16, 16, 160, false, true><<<N_NODES, 256, 0, stream>>>(bufB, nullptr, nullptr, wt + W_RB2, ref_b2, outp, outp);
}
